// Mamba_v2_12240656793951
// MI455X (gfx1250) — matrix operand order repaired and checked by reading, not compiled, not run on hardware
//
#include <hip/hip_runtime.h>
#include <hip/hip_bf16.h>

typedef _Float16 f16;
typedef __attribute__((ext_vector_type(16))) _Float16 v16h;
typedef __attribute__((ext_vector_type(8)))  float    v8f;

union Frag16 { v16h v; unsigned int u[8]; };
union AccF   { v8f  v; float f[8]; };

#define B_SZ      4
#define L_SZ      4096
#define C_SZ      256
#define M_TOT     (B_SZ * L_SZ)      // 16384
#define D_IN      512
#define N_ST      16

// ---------------------------------------------------------------------------
// CDNA5 async copy global->LDS (ASYNCcnt path), per CDNA5 ISA
// ---------------------------------------------------------------------------
__device__ __forceinline__ void async_copy_b128(unsigned lds_addr, const void* gptr) {
    asm volatile("global_load_async_to_lds_b128 %0, %1, off"
                 :: "v"(lds_addr), "v"((unsigned long long)(uintptr_t)gptr)
                 : "memory");
}
__device__ __forceinline__ void wait_async0() {
    asm volatile("s_wait_asynccnt 0x0" ::: "memory");
}

// ---------------------------------------------------------------------------
// 1. Modulation MLP: mods[b, j] = F_clip_s[b,:] . fs_w[j,:] + fs_b[j]
// ---------------------------------------------------------------------------
__global__ void mods_kernel(const float* __restrict__ clip,
                            const float* __restrict__ fsw,
                            const float* __restrict__ fsb,
                            float* __restrict__ mods) {
    int t = blockIdx.x * blockDim.x + threadIdx.x;
    if (t >= B_SZ * 1536) return;
    int b = t / 1536, j = t % 1536;
    const float* cr = clip + b * 512;
    const float* wr = fsw + j * 512;
    float s = fsb[j];
    for (int k = 0; k < 512; ++k) s += cr[k] * wr[k];
    mods[t] = s;
}

// ---------------------------------------------------------------------------
// 2. Tiled transpose  F_content [b, C, L]  ->  x [b, L, C]   (f32)
// ---------------------------------------------------------------------------
__global__ __launch_bounds__(256) void transpose_kernel(const float* __restrict__ src,
                                                        float* __restrict__ dst) {
    __shared__ float tile[32][33];
    int b  = blockIdx.z;
    int c0 = blockIdx.y * 32;
    int l0 = blockIdx.x * 32;
    int tx = threadIdx.x;           // 0..31
    int ty = threadIdx.y;           // 0..7
#pragma unroll
    for (int i = 0; i < 32; i += 8)
        tile[ty + i][tx] = src[((size_t)b * C_SZ + c0 + ty + i) * L_SZ + l0 + tx];
    __syncthreads();
#pragma unroll
    for (int i = 0; i < 32; i += 8)
        dst[((size_t)b * L_SZ + l0 + ty + i) * C_SZ + c0 + tx] = tile[tx][ty + i];
}

// ---------------------------------------------------------------------------
// 3. LayerNorm + AdaLN modulate -> f16  (one block per row of 256)
// ---------------------------------------------------------------------------
__global__ __launch_bounds__(256) void ln_mod_kernel(const float* __restrict__ x,
                                                     const float* __restrict__ mods,
                                                     int shift_off, int scale_off,
                                                     f16* __restrict__ out) {
    int row = blockIdx.x;                 // 0..16383
    int b   = row >> 12;
    int c   = threadIdx.x;                // 0..255
    float v = x[(size_t)row * C_SZ + c];

    __shared__ float red[8];
    float s = v;
#pragma unroll
    for (int off = 16; off > 0; off >>= 1) s += __shfl_down(s, off, 32);
    if ((c & 31) == 0) red[c >> 5] = s;
    __syncthreads();
    float mu = 0.f;
#pragma unroll
    for (int i = 0; i < 8; ++i) mu += red[i];
    mu *= (1.0f / 256.0f);
    __syncthreads();

    float d = v - mu;
    float q = d * d;
#pragma unroll
    for (int off = 16; off > 0; off >>= 1) q += __shfl_down(q, off, 32);
    if ((c & 31) == 0) red[c >> 5] = q;
    __syncthreads();
    float var = 0.f;
#pragma unroll
    for (int i = 0; i < 8; ++i) var += red[i];
    var *= (1.0f / 256.0f);

    float rs = rsqrtf(var + 1e-6f);
    float sc = mods[b * 1536 + scale_off + c];
    float sh = mods[b * 1536 + shift_off + c];
    out[(size_t)row * C_SZ + c] = (f16)((d * rs) * (1.0f + sc) + sh);
}

// ---------------------------------------------------------------------------
// 4. Weight convert f32 [N,K] -> f16 [Npad,Kpad] (zero pad)
// ---------------------------------------------------------------------------
__global__ void wconv_kernel(const float* __restrict__ src, f16* __restrict__ dst,
                             int N, int K, int Npad, int Kpad) {
    int t = blockIdx.x * blockDim.x + threadIdx.x;
    if (t >= Npad * Kpad) return;
    int n = t / Kpad, k = t % Kpad;
    dst[t] = (n < N && k < K) ? (f16)src[n * K + k] : (f16)0.0f;
}

// ---------------------------------------------------------------------------
// 5. WMMA GEMM:  C[M,N] = A[M,K](f16) * W[N,K](f16)^T  (+bias)
//    Block tile 128x64, 8 waves = 4(M) x 2(N), each wave owns 2x2 WMMA tiles.
//    BK=32, double-buffered LDS fed by global_load_async_to_lds_b128.
//    Requires: M % 128 == 0, N % 64 == 0, K % 32 == 0 (caller pads).
//    out_mode 0: C[m*ldc+n]; out_mode 1: C[(b*N+n)*4096 + (m%4096)]
// ---------------------------------------------------------------------------
#define BM 128
#define BN 64
#define BK 32

__global__ __launch_bounds__(256) void gemm_wmma_kernel(
    const f16* __restrict__ A, int lda,
    const f16* __restrict__ W,            // ldw == K
    const float* __restrict__ bias,       // may be null
    float* __restrict__ C, int ldc,
    int M, int N, int K, int out_mode) {

    __shared__ __align__(16) f16 sA[2][BM][BK];   // 2 x 8 KB
    __shared__ __align__(16) f16 sW[2][BN][BK];   // 2 x 4 KB

    const int tid  = threadIdx.x;
    const int wave = tid >> 5;
    const int lane = tid & 31;
    const int wm   = wave & 3;          // 0..3  -> M rows wm*32..wm*32+31
    const int wn   = wave >> 2;         // 0..1  -> N cols wn*32..wn*32+31
    const int bm   = blockIdx.y * BM;
    const int bn   = blockIdx.x * BN;

    // staging coordinates: every thread moves 3 x b128 per K-step (no guards)
    const int st_row = tid >> 2;            // 0..63
    const int st_col = (tid & 3) << 3;      // 0,8,16,24 (halfs)

    AccF acc[2][2];
#pragma unroll
    for (int i = 0; i < 2; ++i)
#pragma unroll
        for (int j = 0; j < 2; ++j)
#pragma unroll
            for (int v = 0; v < 8; ++v) acc[i][j].f[v] = 0.0f;

    const f16* Ab = A + (size_t)bm * lda;
    const f16* Wb = W + (size_t)bn * K;

    auto issue = [&](int b, int k0) {
        async_copy_b128((unsigned)(uintptr_t)&sA[b][st_row][st_col],
                        Ab + (size_t)st_row * lda + k0 + st_col);
        async_copy_b128((unsigned)(uintptr_t)&sA[b][st_row + 64][st_col],
                        Ab + (size_t)(st_row + 64) * lda + k0 + st_col);
        async_copy_b128((unsigned)(uintptr_t)&sW[b][st_row][st_col],
                        Wb + (size_t)st_row * K + k0 + st_col);
    };

    issue(0, 0);

    const int r  = lane & 15;
    const int hi = lane >> 4;
    int buf = 0;

    for (int k0 = 0; k0 < K; k0 += BK) {
        wait_async0();          // this wave's pending async LDS writes done
        __syncthreads();        // all waves' writes visible
        if (k0 + BK < K) issue(buf ^ 1, k0 + BK);

        Frag16 fa[2], fb[2];
#pragma unroll
        for (int i = 0; i < 8; ++i) {
            // A 16x32 f16 layout: lanes 0-15 K{0..7,16..23}, lanes 16-31 K{8..15,24..31}
            int ka = ((i & 4) << 2) + (hi << 3) + ((i & 3) << 1);
            // B 32x16 f16 layout: lanes 0-15 K{0..7,16..23}, lanes 16-31 K{8..15,24..31}
            int kb = ((i & 4) << 2) + (hi << 3) + ((i & 3) << 1);
            fa[0].u[i] = *(const unsigned int*)&sA[buf][wm * 32 + r][ka];
            fa[1].u[i] = *(const unsigned int*)&sA[buf][wm * 32 + 16 + r][ka];
            fb[0].u[i] = *(const unsigned int*)&sW[buf][wn * 32 + r][kb];
            fb[1].u[i] = *(const unsigned int*)&sW[buf][wn * 32 + 16 + r][kb];
        }
        acc[0][0].v = __builtin_amdgcn_wmma_f32_16x16x32_f16(
            false, fa[0].v, false, fb[0].v, (short)0, acc[0][0].v, false, false);
        acc[0][1].v = __builtin_amdgcn_wmma_f32_16x16x32_f16(
            false, fa[0].v, false, fb[1].v, (short)0, acc[0][1].v, false, false);
        acc[1][0].v = __builtin_amdgcn_wmma_f32_16x16x32_f16(
            false, fa[1].v, false, fb[0].v, (short)0, acc[1][0].v, false, false);
        acc[1][1].v = __builtin_amdgcn_wmma_f32_16x16x32_f16(
            false, fa[1].v, false, fb[1].v, (short)0, acc[1][1].v, false, false);
        buf ^= 1;
    }

    const int nl = lane & 15;
#pragma unroll
    for (int ti = 0; ti < 2; ++ti) {
#pragma unroll
        for (int tj = 0; tj < 2; ++tj) {
#pragma unroll
            for (int v = 0; v < 8; ++v) {
                int m = bm + wm * 32 + ti * 16 + v + (hi << 3);
                int n = bn + wn * 32 + tj * 16 + nl;
                float val = acc[ti][tj].f[v] + (bias ? bias[n] : 0.0f);
                if (out_mode == 0) {
                    C[(size_t)m * ldc + n] = val;
                } else {
                    int b = m >> 12, l = m & 4095;
                    C[((size_t)b * N + n) * 4096 + l] = val;
                }
            }
        }
    }
}

// ---------------------------------------------------------------------------
// 6. Causal depthwise conv1d (k=4) + bias + SiLU over xc half of xz
// ---------------------------------------------------------------------------
__global__ void conv_silu_kernel(const float* __restrict__ xz,
                                 const float* __restrict__ cw,
                                 const float* __restrict__ cb,
                                 float* __restrict__ xconv,
                                 f16* __restrict__ xconv16) {
    int t = blockIdx.x * blockDim.x + threadIdx.x;
    if (t >= M_TOT * D_IN) return;
    int d = t & (D_IN - 1);
    int m = t >> 9;
    int l = m & (L_SZ - 1);
    float s = cb[d];
#pragma unroll
    for (int j = 0; j < 4; ++j) {
        int ll = l - 3 + j;
        if (ll >= 0) s += xz[(size_t)(m - 3 + j) * 1024 + d] * cw[d * 4 + j];
    }
    float o = s * (1.0f / (1.0f + expf(-s)));
    xconv[t]   = o;
    xconv16[t] = (f16)o;
}

// ---------------------------------------------------------------------------
// 7. Extract dt_rank cols -> f16 padded [M,32]
// ---------------------------------------------------------------------------
__global__ void dtr_kernel(const float* __restrict__ xdbl, f16* __restrict__ dtr) {
    int t = blockIdx.x * blockDim.x + threadIdx.x;
    if (t >= M_TOT * 32) return;
    int m = t >> 5, k = t & 31;
    dtr[t] = (k < 16) ? (f16)xdbl[(size_t)m * 64 + k] : (f16)0.0f;
}

// ---------------------------------------------------------------------------
// 8. softplus in place
// ---------------------------------------------------------------------------
__global__ void softplus_kernel(float* __restrict__ dt, int n) {
    int t = blockIdx.x * blockDim.x + threadIdx.x;
    if (t >= n) return;
    float v = dt[t];
    dt[t] = (v > 20.0f) ? v : log1pf(expf(v));
}

// ---------------------------------------------------------------------------
// 9. Selective scan: one thread per (b, d); h[16] in registers; B/C in LDS
// ---------------------------------------------------------------------------
__global__ __launch_bounds__(128) void scan_kernel(const float* __restrict__ xconv,
                                                   const float* __restrict__ dt,
                                                   const float* __restrict__ xdbl,
                                                   const float* __restrict__ A_log,
                                                   float* __restrict__ y) {
    int b = blockIdx.y;
    int d = blockIdx.x * 128 + threadIdx.x;     // 0..511
    __shared__ float sBC[32];
    float A[N_ST], h[N_ST];
#pragma unroll
    for (int n = 0; n < N_ST; ++n) {
        A[n] = -expf(A_log[d * N_ST + n]);
        h[n] = 0.0f;
    }
    for (int l = 0; l < L_SZ; ++l) {
        size_t row = (size_t)b * L_SZ + l;
        if (threadIdx.x < 32) sBC[threadIdx.x] = xdbl[row * 64 + 16 + threadIdx.x];
        __syncthreads();
        float xt  = xconv[row * D_IN + d];
        float dtt = dt[row * D_IN + d];
        float acc = 0.0f;
#pragma unroll
        for (int n = 0; n < N_ST; ++n) {
            float dA = expf(dtt * A[n]);
            h[n] = dA * h[n] + dtt * sBC[n] * xt;
            acc += h[n] * sBC[16 + n];
        }
        y[row * D_IN + d] = acc;
        __syncthreads();
    }
}

// ---------------------------------------------------------------------------
// 10. y = (y + xconv*D) * silu(z)  -> f16
// ---------------------------------------------------------------------------
__global__ void gate_kernel(const float* __restrict__ y,
                            const float* __restrict__ xconv,
                            const float* __restrict__ Dp,
                            const float* __restrict__ xz,
                            f16* __restrict__ y16) {
    int t = blockIdx.x * blockDim.x + threadIdx.x;
    if (t >= M_TOT * D_IN) return;
    int d = t & (D_IN - 1);
    int m = t >> 9;
    float v = y[t] + xconv[t] * Dp[d];
    float z = xz[(size_t)m * 1024 + 512 + d];
    float sz = z * (1.0f / (1.0f + expf(-z)));
    y16[t] = (f16)(v * sz);
}

// ---------------------------------------------------------------------------
// 11. x += gate * t     (gate = mods[b, g_off + c])
// ---------------------------------------------------------------------------
__global__ void residual_kernel(float* __restrict__ x, const float* __restrict__ t32,
                                const float* __restrict__ mods, int g_off) {
    int t = blockIdx.x * blockDim.x + threadIdx.x;
    if (t >= M_TOT * C_SZ) return;
    int c = t & (C_SZ - 1);
    int b = t >> 20;                 // t / (4096*256)
    x[t] += mods[b * 1536 + g_off + c] * t32[t];
}

// ---------------------------------------------------------------------------
// 12. gelu(tanh) -> f16
// ---------------------------------------------------------------------------
__global__ void gelu_f16_kernel(const float* __restrict__ src, f16* __restrict__ dst, int n) {
    int t = blockIdx.x * blockDim.x + threadIdx.x;
    if (t >= n) return;
    float v = src[t];
    float u = 0.7978845608028654f * (v + 0.044715f * v * v * v);
    dst[t] = (f16)(0.5f * v * (1.0f + tanhf(u)));
}

// ---------------------------------------------------------------------------
// 13. f32 -> f16
// ---------------------------------------------------------------------------
__global__ void f32_to_f16_kernel(const float* __restrict__ src, f16* __restrict__ dst, int n) {
    int t = blockIdx.x * blockDim.x + threadIdx.x;
    if (t >= n) return;
    dst[t] = (f16)src[t];
}

// ===========================================================================
extern "C" void kernel_launch(void* const* d_in, const int* in_sizes, int n_in,
                              void* d_out, int out_size, void* d_ws, size_t ws_size,
                              hipStream_t stream) {
    const float* F_clip   = (const float*)d_in[0];
    const float* F_cont   = (const float*)d_in[1];
    const float* fs_w     = (const float*)d_in[2];
    const float* fs_b     = (const float*)d_in[3];
    const float* in_pw    = (const float*)d_in[4];
    const float* conv_w   = (const float*)d_in[5];
    const float* conv_b   = (const float*)d_in[6];
    const float* x_pw     = (const float*)d_in[7];
    const float* dt_pw    = (const float*)d_in[8];
    const float* dt_pb    = (const float*)d_in[9];
    const float* A_log    = (const float*)d_in[10];
    const float* D_param  = (const float*)d_in[11];
    const float* out_pw   = (const float*)d_in[12];
    const float* fc1_w    = (const float*)d_in[13];
    const float* fc1_b    = (const float*)d_in[14];
    const float* fc2_w    = (const float*)d_in[15];
    const float* fc2_b    = (const float*)d_in[16];
    const float* co_w     = (const float*)d_in[17];
    const float* co_b     = (const float*)d_in[18];
    float* out = (float*)d_out;

    // ---- workspace layout (256-aligned) ----
    char* ws = (char*)d_ws;
    size_t off = 0;
    auto take = [&](size_t bytes) { size_t o = off; off += (bytes + 255) & ~(size_t)255; return o; };
    float* x      = (float*)(ws + take((size_t)M_TOT * C_SZ  * 4));   // residual stream
    float* mods   = (float*)(ws + take((size_t)B_SZ * 1536   * 4));
    f16*   act16  = (f16*)  (ws + take((size_t)M_TOT * 1024  * 2));   // shared f16 activations
    float* xz     = (float*)(ws + take((size_t)M_TOT * 1024  * 4));
    float* xconv  = (float*)(ws + take((size_t)M_TOT * D_IN  * 4));
    float* xdbl   = (float*)(ws + take((size_t)M_TOT * 64    * 4));
    float* dtbuf  = (float*)(ws + take((size_t)M_TOT * D_IN  * 4));
    float* ybuf   = (float*)(ws + take((size_t)M_TOT * D_IN  * 4));
    float* tmp32  = (float*)(ws + take((size_t)M_TOT * C_SZ  * 4));
    f16*   dtr16  = (f16*)  (ws + take((size_t)M_TOT * 32    * 2));
    f16*   wbuf   = (f16*)  (ws + take((size_t)1024  * 512   * 2));
    (void)ws_size; (void)in_sizes; (void)n_in; (void)out_size;

    const int TPB = 256;
    auto blocks = [&](long n) { return (int)((n + TPB - 1) / TPB); };

    auto gemm = [&](const f16* A, int lda, const float* srcW, int N, int K,
                    int Npad, int Kpad, const float* bias, float* C, int ldc, int mode) {
        wconv_kernel<<<blocks((long)Npad * Kpad), TPB, 0, stream>>>(srcW, wbuf, N, K, Npad, Kpad);
        dim3 grid(Npad / BN, M_TOT / BM);
        gemm_wmma_kernel<<<grid, TPB, 0, stream>>>(A, lda, wbuf, bias, C, ldc,
                                                   M_TOT, Npad, Kpad, mode);
    };

    // 1. modulation vectors
    mods_kernel<<<blocks(B_SZ * 1536), TPB, 0, stream>>>(F_clip, fs_w, fs_b, mods);
    // 2. [b,C,L] -> [b,L,C]
    transpose_kernel<<<dim3(L_SZ / 32, C_SZ / 32, B_SZ), dim3(32, 8), 0, stream>>>(F_cont, x);
    // 3. LN + modulate (msa)
    ln_mod_kernel<<<M_TOT, 256, 0, stream>>>(x, mods, 0, 256, act16);
    // 4. in_proj: [M,256] x [1024,256]^T -> xz [M,1024]
    gemm(act16, 256, in_pw, 1024, 256, 1024, 256, nullptr, xz, 1024, 0);
    // 5. depthwise causal conv + SiLU
    conv_silu_kernel<<<blocks((long)M_TOT * D_IN), TPB, 0, stream>>>(xz, conv_w, conv_b, xconv, act16);
    // 6. x_proj: [M,512] x [48,512]^T -> xdbl [M,64] (padded)
    gemm(act16, 512, x_pw, 48, 512, 64, 512, nullptr, xdbl, 64, 0);
    // 7. dt low-rank -> f16 padded
    dtr_kernel<<<blocks((long)M_TOT * 32), TPB, 0, stream>>>(xdbl, dtr16);
    // 8. dt_proj + bias, softplus
    gemm(dtr16, 32, dt_pw, 512, 16, 512, 32, dt_pb, dtbuf, 512, 0);
    softplus_kernel<<<blocks((long)M_TOT * D_IN), TPB, 0, stream>>>(dtbuf, M_TOT * D_IN);
    // 9. selective scan
    scan_kernel<<<dim3(D_IN / 128, B_SZ), 128, 0, stream>>>(xconv, dtbuf, xdbl, A_log, ybuf);
    // 10. gate
    gate_kernel<<<blocks((long)M_TOT * D_IN), TPB, 0, stream>>>(ybuf, xconv, D_param, xz, act16);
    // 11. out_proj -> tmp32 [M,256]
    gemm(act16, 512, out_pw, 256, 512, 256, 512, nullptr, tmp32, 256, 0);
    // 12. x += g_msa * mamba_out
    residual_kernel<<<blocks((long)M_TOT * C_SZ), TPB, 0, stream>>>(x, tmp32, mods, 512);
    // 13. LN + modulate (mlp)
    ln_mod_kernel<<<M_TOT, 256, 0, stream>>>(x, mods, 768, 1024, act16);
    // 14. fc1 + bias -> tmp32; gelu -> act16
    gemm(act16, 256, fc1_w, 256, 256, 256, 256, fc1_b, tmp32, 256, 0);
    gelu_f16_kernel<<<blocks((long)M_TOT * C_SZ), TPB, 0, stream>>>(tmp32, act16, M_TOT * C_SZ);
    // 15. fc2 + bias -> tmp32
    gemm(act16, 256, fc2_w, 256, 256, 256, 256, fc2_b, tmp32, 256, 0);
    // 16. x += g_mlp * m
    residual_kernel<<<blocks((long)M_TOT * C_SZ), TPB, 0, stream>>>(x, tmp32, mods, 1280);
    // 17. final 1x1 conv with transposed store -> d_out [b, 256, 4096]
    f32_to_f16_kernel<<<blocks((long)M_TOT * C_SZ), TPB, 0, stream>>>(x, act16, M_TOT * C_SZ);
    gemm(act16, 256, co_w, 256, 256, 256, 256, co_b, out, 256, 1);
}
